// QNet_12317966204960
// MI455X (gfx1250) — matrix operand order repaired and checked by reading, not compiled, not run on hardware
//
#include <hip/hip_runtime.h>
#include <math.h>

// ---------------------------------------------------------------------------
// Problem constants (match reference)
// ---------------------------------------------------------------------------
#define BB 2
#define LL 1024
#define VOCAB 32000
#define DD 512
#define NL 2
#define INNER 1024          // 2*D
#define MLPD 2048
#define NSEG 8
#define SEG 128             // L / NSEG
#define NSEQ (BB * NSEG)    // 16 subsequences

typedef __attribute__((ext_vector_type(16))) _Float16 v16h;
typedef __attribute__((ext_vector_type(8)))  float    v8f;
typedef __attribute__((ext_vector_type(4)))  float    v4f;

// ---------------------------------------------------------------------------
// WMMA GEMM:  C[M,N] (+)= act( A[M,K](f32, lda) * Bt[N,K](f16) ) (+bias)
//   - wave computes MT x 4 tiles of 16x16  (MT=2 -> 32 rows x 64 cols)
//   - 8 waves arranged 2(M) x 4(N) -> block tile (32*MT) x 256
//   - register double-buffered K pipeline (K % 64 == 0 required)
//   - STORE_MODE: 0 = store, 1 = accumulate into C (+=), 2 = SiLU then store
//   - SEG_LIMIT:  blockIdx.z = subsequence; skip row-tiles >= (i+1)*SEG
//   - NTA:        non-temporal loads for streaming A operand
// Lane layouts used by this code (contiguous k order, same in A and B):
//   A (16x32 f16):  row = lane&15 ; K = kbA + e,               kbA = lane<16?0:16
//   B (32x16 f16):  col = lane&15 ; K = kbB + e,               kbB = lane<16?0:16
//   C (16x16 f32):  col = lane&15 ; row = (lane<16?0:8) + vgpr_j
// ---------------------------------------------------------------------------
template<int MT, int STORE_MODE, bool BIAS, bool SEG_LIMIT, bool NTA>
__global__ __launch_bounds__(256)
void gemm_wmma(const float* __restrict__ A, const _Float16* __restrict__ Bt,
               float* __restrict__ C, const float* __restrict__ bias,
               int M, int N, int K, int lda, int ldc,
               long long strideA, long long strideC)
{
    constexpr int NT = 4;
    const int lane  = threadIdx.x & 31;
    const int wave  = threadIdx.x >> 5;
    const int waveM = wave >> 2;            // 0..1
    const int waveN = wave & 3;             // 0..3
    const int z     = blockIdx.z;

    if (SEG_LIMIT) {
        const int Li = ((z & (NSEG - 1)) + 1) * SEG;   // valid rows for this subsequence
        if ((int)(blockIdx.y * (32 * MT)) >= Li) return;
    }

    const int m0 = blockIdx.y * (32 * MT) + waveM * (16 * MT);
    if (m0 >= M) return;
    const int n0 = (blockIdx.x * 4 + waveN) * (16 * NT);
    if (n0 >= N) return;

    const int rowA = lane & 15;
    const int kbA  = (lane < 16) ? 0 : 16;
    const int colB = lane & 15;
    const int kbB  = (lane < 16) ? 0 : 16;

    const float* ApR[MT];
#pragma unroll
    for (int mi = 0; mi < MT; ++mi)
        ApR[mi] = A + (size_t)z * strideA + (size_t)(m0 + mi * 16 + rowA) * lda + kbA;
    const _Float16* BpR[NT];
#pragma unroll
    for (int t = 0; t < NT; ++t)
        BpR[t] = Bt + (size_t)(n0 + t * 16 + colB) * K + kbB;

    v8f acc[MT][NT];
#pragma unroll
    for (int mi = 0; mi < MT; ++mi)
#pragma unroll
        for (int t = 0; t < NT; ++t)
            acc[mi][t] = (v8f){0.f,0.f,0.f,0.f,0.f,0.f,0.f,0.f};

    auto loadA = [&](v4f (&dst)[MT][4], int k) {
#pragma unroll
        for (int mi = 0; mi < MT; ++mi) {
            const v4f* p = (const v4f*)(ApR[mi] + k);
            const v4f* q = (const v4f*)(ApR[mi] + k + 8);
            if (NTA) {
                dst[mi][0] = __builtin_nontemporal_load(p);
                dst[mi][1] = __builtin_nontemporal_load(p + 1);
                dst[mi][2] = __builtin_nontemporal_load(q);
                dst[mi][3] = __builtin_nontemporal_load(q + 1);
            } else {
                dst[mi][0] = p[0]; dst[mi][1] = p[1];
                dst[mi][2] = q[0]; dst[mi][3] = q[1];
            }
        }
    };
    auto loadB = [&](v16h (&dst)[NT], int k) {
#pragma unroll
        for (int t = 0; t < NT; ++t)
            dst[t] = *(const v16h*)(BpR[t] + k);     // 32 contiguous bytes
    };
    auto mmastep = [&](v4f (&ar)[MT][4], v16h (&b)[NT]) {
        v16h af[MT];
#pragma unroll
        for (int mi = 0; mi < MT; ++mi)
#pragma unroll
            for (int i = 0; i < 4; ++i)
#pragma unroll
                for (int j = 0; j < 4; ++j)
                    af[mi][4 * i + j] = (_Float16)ar[mi][i][j];
#pragma unroll
        for (int mi = 0; mi < MT; ++mi)
#pragma unroll
            for (int t = 0; t < NT; ++t)
                acc[mi][t] = __builtin_amdgcn_wmma_f32_16x16x32_f16(
                    false, af[mi], false, b[t], (short)0, acc[mi][t], false, false);
    };

    // ---- double-buffered K pipeline (K % 64 == 0) ----
    v4f a0r[MT][4], a1r[MT][4];
    v16h b0[NT], b1[NT];
    loadA(a0r, 0); loadB(b0, 0);
    for (int k0 = 0; k0 < K; k0 += 64) {
        loadA(a1r, k0 + 32); loadB(b1, k0 + 32);
        mmastep(a0r, b0);
        if (k0 + 64 < K) { loadA(a0r, k0 + 64); loadB(b0, k0 + 64); }
        mmastep(a1r, b1);
    }

    // ---- epilogue ----
    const int crow = (lane < 16) ? 0 : 8;
    const int ccol = lane & 15;
#pragma unroll
    for (int mi = 0; mi < MT; ++mi) {
#pragma unroll
        for (int t = 0; t < NT; ++t) {
            const int n = n0 + t * 16 + ccol;
            const float bv = BIAS ? bias[n] : 0.f;
            float* cp = C + (size_t)z * strideC + (size_t)(m0 + mi * 16 + crow) * ldc + n;
#pragma unroll
            for (int j = 0; j < 8; ++j) {
                float v = acc[mi][t][j];
                if (STORE_MODE == 2) v = v / (1.f + __expf(-v));   // SiLU
                if (BIAS) v += bv;
                float* dst = cp + (size_t)j * ldc;
                if (STORE_MODE == 1) *dst += v;
                else                 *dst  = v;
            }
        }
    }
}

// ---------------------------------------------------------------------------
// fp32 [R][C]  ->  f16 transposed [C][R]   (coalesced writes)
// ---------------------------------------------------------------------------
__global__ void transpose_cvt(const float* __restrict__ in, _Float16* __restrict__ out,
                              int R, int C)
{
    size_t tid = (size_t)blockIdx.x * 256 + threadIdx.x;
    if (tid >= (size_t)R * C) return;
    int c = (int)(tid / R);
    int r = (int)(tid % R);
    out[tid] = (_Float16)in[(size_t)r * C + c];
}

// correct[row, :] = E[input_ids[row], :]
__global__ void gather_correct(const int* __restrict__ ids, const float* __restrict__ E,
                               float* __restrict__ out)
{
    int tid = blockIdx.x * 256 + threadIdx.x;          // BB*LL*DD threads
    int row = tid >> 9, d = tid & 511;
    out[tid] = E[(size_t)ids[row] * DD + d];
}

// x[s,t,:] = (t < i*SEG) ? correct[b,t,:] : dec[b,t,:]   (only for t < (i+1)*SEG)
__global__ void build_x(const float* __restrict__ correct, const float* __restrict__ dec,
                        float* __restrict__ x)
{
    const int s = blockIdx.z, t = blockIdx.y;
    const int d = blockIdx.x * 256 + threadIdx.x;
    const int i = s & (NSEG - 1), b = s >> 3;
    if (t >= (i + 1) * SEG) return;
    const size_t src = ((size_t)(b * LL + t)) * DD + d;
    const float v = (t < i * SEG) ? correct[src] : dec[src];
    x[((size_t)s * LL + t) * DD + d] = v;
}

// Fused linear recurrence + gate:  per (seq, channel):
//   h_t = a*h_{t-1} + u_t ;  u-slot <- h_t * silu(g_t)
// ug layout: [NSEQ][LL][2*INNER], u = cols [0,INNER), g = cols [INNER,2*INNER)
__global__ void recur_gate(float* __restrict__ ug, const float* __restrict__ Adec)
{
    const int tid = blockIdx.x * 256 + threadIdx.x;    // NSEQ*INNER threads
    const int s = tid >> 10, c = tid & (INNER - 1);
    const int Li = ((s & (NSEG - 1)) + 1) * SEG;
    const float a = Adec[c];
    float h = 0.f;
    float* base = ug + (size_t)s * LL * (2 * INNER);
    for (int t = 0; t < Li; ++t) {
        const float u = base[(size_t)t * (2 * INNER) + c];
        const float g = base[(size_t)t * (2 * INNER) + INNER + c];
        h = fmaf(a, h, u);
        base[(size_t)t * (2 * INNER) + c] = h * (g / (1.f + __expf(-g)));
    }
}

// last[s,:] = x[s, (i+1)*SEG - 1, :]
__global__ void gather_last(const float* __restrict__ x, float* __restrict__ last)
{
    int tid = blockIdx.x * 256 + threadIdx.x;          // NSEQ*DD threads
    int s = tid >> 9, d = tid & 511;
    int Li = ((s & (NSEG - 1)) + 1) * SEG;
    last[tid] = x[((size_t)s * LL + (Li - 1)) * DD + d];
}

// ---------------------------------------------------------------------------
extern "C" void kernel_launch(void* const* d_in, const int* in_sizes, int n_in,
                              void* d_out, int out_size, void* d_ws, size_t ws_size,
                              hipStream_t stream)
{
    (void)in_sizes; (void)n_in; (void)out_size; (void)ws_size;

    const float* decoder = (const float*)d_in[0];   // (B,L,VOCAB)
    const int*   ids     = (const int*)  d_in[1];   // (B,L,1)
    /* d_in[2] segmentation indices: fixed starts i*SEG, derived analytically */
    const float* E       = (const float*)d_in[3];   // (VOCAB,D)
    const float* Adecay  = (const float*)d_in[4];   // (NL,INNER)
    const float* Win     = (const float*)d_in[5];   // (NL,D,2*INNER)
    const float* Wout    = (const float*)d_in[6];   // (NL,INNER,D)
    const float* Wm1     = (const float*)d_in[7];   // (NL,D,MLP)
    const float* Wm2     = (const float*)d_in[8];   // (NL,MLP,D)
    const float* muW     = (const float*)d_in[9];
    const float* mub     = (const float*)d_in[10];
    const float* lvW     = (const float*)d_in[11];
    const float* lvb     = (const float*)d_in[12];
    float* out = (float*)d_out;                     // mu (16*512) ++ logvar (16*512)

    // ---- workspace carve-up ----
    char* w = (char*)d_ws;
    auto alloc = [&](size_t bytes) { char* p = w; w += (bytes + 255) & ~(size_t)255; return p; };
    _Float16* Et    = (_Float16*)alloc((size_t)DD * VOCAB * 2);
    _Float16* Wint  = (_Float16*)alloc((size_t)NL * (2*INNER) * DD * 2);
    _Float16* Woutt = (_Float16*)alloc((size_t)NL * DD * INNER * 2);
    _Float16* Wm1t  = (_Float16*)alloc((size_t)NL * MLPD * DD * 2);
    _Float16* Wm2t  = (_Float16*)alloc((size_t)NL * DD * MLPD * 2);
    _Float16* muWt  = (_Float16*)alloc((size_t)DD * DD * 2);
    _Float16* lvWt  = (_Float16*)alloc((size_t)DD * DD * 2);
    float* correct  = (float*)alloc((size_t)BB * LL * DD * 4);
    float* dec      = (float*)alloc((size_t)BB * LL * DD * 4);
    float* x        = (float*)alloc((size_t)NSEQ * LL * DD * 4);
    float* ug       = (float*)alloc((size_t)NSEQ * LL * (2*INNER) * 4);
    float* last     = (float*)alloc((size_t)NSEQ * DD * 4);

    auto tcvt = [&](const float* in, _Float16* o, int R, int C) {
        size_t n = (size_t)R * C;
        transpose_cvt<<<dim3((unsigned)((n + 255) / 256)), 256, 0, stream>>>(in, o, R, C);
    };

    // ---- 1. weight prep (one-time transposes to f16 [N][K]) ----
    tcvt(E, Et, VOCAB, DD);                                         // Et: [D][VOCAB]
    for (int l = 0; l < NL; ++l) {
        tcvt(Win  + (size_t)l * DD * (2*INNER), Wint  + (size_t)l * (2*INNER) * DD, DD,    2*INNER);
        tcvt(Wout + (size_t)l * INNER * DD,     Woutt + (size_t)l * DD * INNER,     INNER, DD);
        tcvt(Wm1  + (size_t)l * DD * MLPD,      Wm1t  + (size_t)l * MLPD * DD,      DD,    MLPD);
        tcvt(Wm2  + (size_t)l * MLPD * DD,      Wm2t  + (size_t)l * DD * MLPD,      MLPD,  DD);
    }
    tcvt(muW, muWt, DD, DD);
    tcvt(lvW, lvWt, DD, DD);

    // ---- 2. ground-truth embeddings + tied projection of logits ----
    gather_correct<<<dim3(BB * LL * DD / 256), 256, 0, stream>>>(ids, E, correct);
    // dec = decoder @ E : M=2048, N=512, K=32000  (A streamed non-temporally)
    gemm_wmma<2, 0, false, false, true><<<dim3(2, 32, 1), 256, 0, stream>>>(
        decoder, Et, dec, nullptr, BB * LL, DD, VOCAB, VOCAB, DD, 0, 0);

    // ---- 3. assemble the 16 subsequences ----
    build_x<<<dim3(DD / 256, LL, NSEQ), 256, 0, stream>>>(correct, dec, x);

    // ---- 4. mixer layers ----
    const long long sX  = (long long)LL * DD;
    const long long sUG = (long long)LL * (2 * INNER);
    for (int l = 0; l < NL; ++l) {
        // ug = x @ Win[l] : K=512, N=2048
        gemm_wmma<2, 0, false, true, false><<<dim3(8, LL / 64, NSEQ), 256, 0, stream>>>(
            x, Wint + (size_t)l * (2*INNER) * DD, ug, nullptr,
            LL, 2 * INNER, DD, DD, 2 * INNER, sX, sUG);
        // h_t = a h + u ; u-slot <- silu(g)*h
        recur_gate<<<dim3(NSEQ * INNER / 256), 256, 0, stream>>>(
            ug, Adecay + (size_t)l * INNER);
        // x += gh @ Wout[l] : K=1024, N=512  (A = ug with lda=2048)
        gemm_wmma<2, 1, false, true, false><<<dim3(2, LL / 64, NSEQ), 256, 0, stream>>>(
            ug, Woutt + (size_t)l * DD * INNER, x, nullptr,
            LL, DD, INNER, 2 * INNER, DD, sUG, sX);
        // m = silu(x @ Wm1[l]) : K=512, N=2048 (stored into ug)
        gemm_wmma<2, 2, false, true, false><<<dim3(8, LL / 64, NSEQ), 256, 0, stream>>>(
            x, Wm1t + (size_t)l * MLPD * DD, ug, nullptr,
            LL, MLPD, DD, DD, 2 * INNER, sX, sUG);
        // x += m @ Wm2[l] : K=2048, N=512
        gemm_wmma<2, 1, false, true, false><<<dim3(2, LL / 64, NSEQ), 256, 0, stream>>>(
            ug, Wm2t + (size_t)l * DD * MLPD, x, nullptr,
            LL, DD, MLPD, 2 * INNER, DD, sUG, sX);
    }

    // ---- 5. heads ----
    gather_last<<<dim3(NSEQ * DD / 256), 256, 0, stream>>>(x, last);
    // mu = last @ mu_W + mu_b : M=16, N=512, K=512
    gemm_wmma<1, 0, true, false, false><<<dim3(2, 1, 1), 256, 0, stream>>>(
        last, muWt, out, mub, NSEQ, DD, DD, DD, DD, 0, 0);
    gemm_wmma<1, 0, true, false, false><<<dim3(2, 1, 1), 256, 0, stream>>>(
        last, lvWt, out + (size_t)NSEQ * DD, lvb, NSEQ, DD, DD, DD, DD, 0, 0);
}
